// ConformerRNNTModel_83202106458620
// MI455X (gfx1250) — matrix operand order repaired and checked by reading, not compiled, not run on hardware
//
#include <hip/hip_runtime.h>

#define B_ 4
#define T_ 256
#define U_ 100
#define E_ 512
#define V_ 1024
#define D_ 512
#define H_ 512
#define U1 101
#define ROWS (B_*U1)      /* 404 valid predictor rows */
#define ROWS_PAD 416      /* 26 row tiles */

typedef __attribute__((ext_vector_type(16))) __bf16 v16bf;
typedef __attribute__((ext_vector_type(8)))  float  v8f;
typedef unsigned int u32x4 __attribute__((ext_vector_type(4)));
typedef int v4i __attribute__((ext_vector_type(4)));

#if defined(__HIP_DEVICE_COMPILE__) && __has_builtin(__builtin_amdgcn_global_load_async_to_lds_b128)
#define HAVE_ASYNC_LDS 1
typedef __attribute__((address_space(1))) v4i g_v4i;   // global int4 (builtin param type)
typedef __attribute__((address_space(3))) v4i l_v4i;   // LDS int4
#else
#define HAVE_ASYNC_LDS 0
#endif

union AFrag { u32x4 q[2]; v16bf v; };

__device__ __forceinline__ unsigned short f2bf(float f) {
  unsigned int x = __float_as_uint(f);
  unsigned int r = (x + 0x7FFFu + ((x >> 16) & 1u)) >> 16;
  return (unsigned short)r;
}
__device__ __forceinline__ float sigf(float x) { return 1.0f / (1.0f + __expf(-x)); }

// A fragment (16x32 bf16 tile) from a row-major bf16 row base.
// Lane L holds row M=L%16; elements 0..7 at K=k0+off, 8..15 at K=k0+off+16,
// off = 0 (lanes 0-15) or 8 (lanes 16-31).  [CDNA5 ISA 16-bit A layout]
__device__ __forceinline__ v16bf load_a_frag(const unsigned short* rowbase, int k0, int lane) {
  int off = (lane < 16) ? 0 : 8;
  AFrag a;
  a.q[0] = *(const u32x4*)(rowbase + k0 + off);
  a.q[1] = *(const u32x4*)(rowbase + k0 + off + 16);
  return a.v;
}
// B fragment (32x16 K-major tile) for C = X @ W^T with W stored [N][K] row-major:
// lane L holds column N = n; elements 0..7 at K=k0+off, 8..15 at K=k0+off+16, off = (L<16 ? 0 : 8).
__device__ __forceinline__ v16bf load_b_frag(const unsigned short* __restrict__ W, int n, int k0, int lane) {
  int kb = k0 + ((lane < 16) ? 0 : 8);
  AFrag b;
  b.q[0] = *(const u32x4*)(W + (size_t)n * 512 + kb);
  b.q[1] = *(const u32x4*)(W + (size_t)n * 512 + kb + 16);
  return b.v;
}

// ---------------- prep kernels ----------------
__global__ void f2bf_kernel(const float* __restrict__ src, unsigned short* __restrict__ dst, int n) {
  for (int i = blockIdx.x * blockDim.x + threadIdx.x; i < n; i += gridDim.x * blockDim.x)
    dst[i] = f2bf(src[i]);
}

__global__ void zerou16_kernel(unsigned short* __restrict__ p, int n) {
  for (int i = blockIdx.x * blockDim.x + threadIdx.x; i < n; i += gridDim.x * blockDim.x)
    p[i] = 0;
}

// xemb padded to 16 "batch" rows so LSTM A-fragment loads need no EXEC guards.
__global__ void embed_kernel(const int* __restrict__ tokens, const float* __restrict__ emb,
                             unsigned short* __restrict__ xemb) {
  const int n = 16 * U1 * D_;
  for (int i = blockIdx.x * blockDim.x + threadIdx.x; i < n; i += gridDim.x * blockDim.x) {
    int d = i & 511;
    int bu = i >> 9;
    int b = bu / U1, u = bu % U1;
    float v = 0.0f;
    if (b < B_) {
      int tok = (u == 0) ? 0 : tokens[b * U_ + (u - 1)];
      v = emb[(size_t)tok * D_ + d];
    }
    xemb[i] = f2bf(v);
  }
}

// ---------------- predictor LSTM: single workgroup, WMMA gates ----------------
__global__ void lstm_kernel(const unsigned short* __restrict__ xemb,
                            const unsigned short* __restrict__ Wih,
                            const unsigned short* __restrict__ Whh,
                            const float* __restrict__ bl,
                            unsigned short* __restrict__ hstates) {
  __shared__ __align__(32) unsigned short hS[16 * 512]; // rows 0..3 = batch, rest stay 0
  __shared__ float cS[4 * 512];
  __shared__ float gS[4 * 2048];
  const int tid = threadIdx.x;
  const int lane = tid & 31, wid = tid >> 5;
  const int M = lane & 15;
  for (int i = tid; i < 16 * 512; i += 256) hS[i] = 0;
  for (int i = tid; i < 4 * 512; i += 256) cS[i] = 0.0f;
  __syncthreads();

  for (int t = 0; t < U1; ++t) {
    const unsigned short* xrow = xemb + ((size_t)(M * U1 + t)) * D_; // padded rows: no guard
    for (int i = 0; i < 16; ++i) {
      int nt = i * 8 + wid;            // 128 gate-column tiles over 8 waves
      int n = nt * 16 + M;
      v8f acc = {0.f, 0.f, 0.f, 0.f, 0.f, 0.f, 0.f, 0.f};
#pragma unroll 4
      for (int kc = 0; kc < 16; ++kc) {
        int k0 = kc * 32;
        v16bf av = load_a_frag(xrow, k0, lane);
        v16bf bv = load_b_frag(Wih, n, k0, lane);
        acc = __builtin_amdgcn_wmma_f32_16x16x32_bf16(false, av, false, bv, (short)0, acc, false, false);
      }
#pragma unroll 4
      for (int kc = 0; kc < 16; ++kc) {
        int k0 = kc * 32;
        v16bf av = load_a_frag(hS + M * 512, k0, lane);
        v16bf bv = load_b_frag(Whh, n, k0, lane);
        acc = __builtin_amdgcn_wmma_f32_16x16x32_bf16(false, av, false, bv, (short)0, acc, false, false);
      }
#pragma unroll
      for (int j = 0; j < 8; ++j) {
        int Mw = j + ((lane >= 16) ? 8 : 0);   // C layout: VGPR j = row j / j+8
        if (Mw < 4) gS[Mw * 2048 + n] = acc[j] + bl[n];
      }
    }
    __syncthreads();
    for (int i = tid; i < 4 * 512; i += 256) {
      int b = i >> 9, hh = i & 511;
      float gi = gS[b * 2048 + hh];
      float gf = gS[b * 2048 + 512 + hh];
      float gg = gS[b * 2048 + 1024 + hh];
      float go = gS[b * 2048 + 1536 + hh];
      float c = sigf(gf) * cS[i] + sigf(gi) * tanhf(gg);
      float h = sigf(go) * tanhf(c);
      cS[i] = c;
      unsigned short hb = f2bf(h);
      hS[b * 512 + hh] = hb;
      hstates[((size_t)(b * U1 + t)) * H_ + hh] = hb;
    }
    __syncthreads();
  }
}

// ---------------- predictor projection: pred_out = h @ W_pred^T + b ----------------
__global__ void pred_kernel(const unsigned short* __restrict__ hstates,
                            const unsigned short* __restrict__ Wpred,
                            const float* __restrict__ bp,
                            float* __restrict__ pred_out) {
  const int tid = threadIdx.x, lane = tid & 31, wid = tid >> 5;
  const int rt = blockIdx.x;          // 26 row tiles (416 padded rows)
  const int M = lane & 15;
  const unsigned short* arow = hstates + (size_t)(rt * 16 + M) * H_;  // padded: no guard
  for (int i = 0; i < 4; ++i) {
    int nt = i * 8 + wid;             // 32 column tiles over 8 waves
    int n = nt * 16 + M;
    v8f acc = {0.f, 0.f, 0.f, 0.f, 0.f, 0.f, 0.f, 0.f};
#pragma unroll 4
    for (int kc = 0; kc < 16; ++kc) {
      int k0 = kc * 32;
      v16bf av = load_a_frag(arow, k0, lane);
      v16bf bv = load_b_frag(Wpred, n, k0, lane);
      acc = __builtin_amdgcn_wmma_f32_16x16x32_bf16(false, av, false, bv, (short)0, acc, false, false);
    }
#pragma unroll
    for (int j = 0; j < 8; ++j) {
      int Mw = j + ((lane >= 16) ? 8 : 0);
      int Rw = rt * 16 + Mw;
      int col = nt * 16 + M;
      if (Rw < ROWS) pred_out[(size_t)Rw * E_ + col] = acc[j] + bp[col];
    }
  }
}

// ---- fused joiner: tanh(enc+pred) @ Wjoin^T + b_join, online logsumexp,
// ---- emits only blank/label log-probs (never materializes [B,T,U+1,V] logits)
__global__ void joiner_kernel(const float* __restrict__ enc,
                              const float* __restrict__ pred_out,
                              const unsigned short* __restrict__ Wjoin,
                              const float* __restrict__ bj,
                              const int* __restrict__ tokens,
                              float* __restrict__ blank_lp,
                              float* __restrict__ label_lp) {
  __shared__ __align__(16) float encS[E_];
  __shared__ __align__(32) unsigned short As[16 * E_];  // tanh(enc+pred) tile, bf16
  __shared__ float scr[8][16][16];                      // per-wave C-tile scratch
  __shared__ float mW[8][16], sW[8][16], blW[8][16], laW[8][16];
#if HAVE_ASYNC_LDS
  __shared__ __align__(16) float predS[16 * E_];        // async-staged pred rows
#endif
  const int tid = threadIdx.x, lane = tid & 31, wid = tid >> 5;
  const int ut = blockIdx.x, t = blockIdx.y, b = blockIdx.z;
  const float* encRow = enc + ((size_t)(b * T_ + t)) * E_;

#if HAVE_ASYNC_LDS
  // Stage enc row (2KB) + pred tile (32KB) via GLOBAL_LOAD_ASYNC_TO_LDS_B128 (ASYNCcnt).
  if (tid < 128)
    __builtin_amdgcn_global_load_async_to_lds_b128(
        (g_v4i*)(encRow + tid * 4), (l_v4i*)(encS + tid * 4), 0, 0);
  for (int c = tid; c < 2048; c += 256) {
    int r = c >> 7, cc = (c & 127) * 4;
    int u = ut * 16 + r;
    if (u < U1)
      __builtin_amdgcn_global_load_async_to_lds_b128(
          (g_v4i*)(pred_out + ((size_t)(b * U1 + u)) * E_ + cc),
          (l_v4i*)(predS + r * E_ + cc), 0, 0);
  }
  asm volatile("s_wait_asynccnt 0x0" ::: "memory");
  __syncthreads();
  for (int i = tid; i < 16 * E_; i += 256) {
    int r = i >> 9, e = i & 511;
    int u = ut * 16 + r;
    float v = 0.0f;
    if (u < U1) v = tanhf(encS[e] + predS[r * E_ + e]);
    As[i] = f2bf(v);
  }
#else
  for (int i = tid; i < E_; i += 256) encS[i] = encRow[i];
  __syncthreads();
  for (int i = tid; i < 16 * E_; i += 256) {
    int r = i >> 9, e = i & 511;
    int u = ut * 16 + r;
    float v = 0.0f;
    if (u < U1) v = tanhf(encS[e] + pred_out[((size_t)(b * U1 + u)) * E_ + e]);
    As[i] = f2bf(v);
  }
#endif
  __syncthreads();

  const int M = lane & 15;
  const int u_r = ut * 16 + lane;                 // row owned by this lane (lane<16)
  int labelCol = -1;
  if (lane < 16 && u_r < U_) labelCol = tokens[b * U_ + u_r];
  float m = -1e30f, s = 0.0f, blv = -1e30f, lav = -1e30f;
  const unsigned short* Arow = As + M * E_;

  for (int i = 0; i < 8; ++i) {
    int vt = i * 8 + wid;                         // 64 V-tiles over 8 waves
    int n = vt * 16 + M;
    float bias = bj[n];
    const unsigned short* Brow = Wjoin + (size_t)n * 512;
    v8f acc = {0.f, 0.f, 0.f, 0.f, 0.f, 0.f, 0.f, 0.f};
    // software-pipelined K loop: prefetch k+1 fragments before WMMA on k
    v16bf av = load_a_frag(Arow, 0, lane);
    v16bf bv = load_b_frag(Brow, 0, 0, lane);
#pragma unroll
    for (int kc = 0; kc < 16; ++kc) {
      v16bf av_n = av, bv_n = bv;
      if (kc < 15) {
        int k1 = (kc + 1) * 32;
        av_n = load_a_frag(Arow, k1, lane);
        bv_n = load_b_frag(Brow, 0, k1, lane);
      }
      acc = __builtin_amdgcn_wmma_f32_16x16x32_bf16(false, av, false, bv, (short)0, acc, false, false);
      av = av_n; bv = bv_n;
    }
#pragma unroll
    for (int j = 0; j < 8; ++j) {
      int Mw = j + ((lane >= 16) ? 8 : 0);
      scr[wid][Mw][M] = acc[j] + bias;
    }
    __syncthreads();
    if (lane < 16) {                              // online logsumexp per row
      float tm = -1e30f;
#pragma unroll
      for (int nn = 0; nn < 16; ++nn) tm = fmaxf(tm, scr[wid][lane][nn]);
      float nm = fmaxf(m, tm);
      float as = s * __expf(m - nm);
#pragma unroll
      for (int nn = 0; nn < 16; ++nn) as += __expf(scr[wid][lane][nn] - nm);
      m = nm; s = as;
      if (vt == 0) blv = scr[wid][lane][0];
      if (labelCol >= 0 && (labelCol >> 4) == vt) lav = scr[wid][lane][labelCol & 15];
    }
    __syncthreads();
  }
  if (lane < 16) { mW[wid][lane] = m; sW[wid][lane] = s; blW[wid][lane] = blv; laW[wid][lane] = lav; }
  __syncthreads();
  if (wid == 0 && lane < 16) {
    float MM = -1e30f;
    for (int w = 0; w < 8; ++w) MM = fmaxf(MM, mW[w][lane]);
    float SS = 0.0f, BL = -1e30f, LA = -1e30f;
    for (int w = 0; w < 8; ++w) {
      SS += sW[w][lane] * __expf(mW[w][lane] - MM);
      BL = fmaxf(BL, blW[w][lane]);
      LA = fmaxf(LA, laW[w][lane]);
    }
    float lse = MM + __logf(SS);
    int u = ut * 16 + lane;
    if (u < U1) blank_lp[((size_t)(b * T_ + t)) * U1 + u] = BL - lse;
    if (u < U_) label_lp[((size_t)(b * T_ + t)) * U_ + u] = LA - lse;
  }
}

// ---------------- RNNT forward DP (Graves alpha recursion) ----------------
__global__ void dp_kernel(const float* __restrict__ blank_lp,
                          const float* __restrict__ label_lp,
                          float* __restrict__ out) {
  __shared__ float alpha[4][U1 + 3];
  __shared__ float llv[4];
  const int b = threadIdx.x;
  if (b < B_) {
    const float* blp = blank_lp + (size_t)b * T_ * U1;
    const float* llp = label_lp + (size_t)b * T_ * U_;
    alpha[b][0] = 0.0f;
    for (int u = 1; u <= U_; ++u) alpha[b][u] = alpha[b][u - 1] + llp[u - 1];
    for (int t = 1; t < T_; ++t) {
      const float* bprev = blp + (size_t)(t - 1) * U1;
      const float* lrow = llp + (size_t)t * U_;
      float carry = alpha[b][0] + bprev[0];
      alpha[b][0] = carry;
      for (int u = 1; u <= U_; ++u) {
        float eu = alpha[b][u] + bprev[u];
        float cu = carry + lrow[u - 1];
        float mx = fmaxf(eu, cu);
        carry = mx + log1pf(expf(fminf(eu, cu) - mx));
        alpha[b][u] = carry;
      }
    }
    llv[b] = alpha[b][U_] + blp[(size_t)(T_ - 1) * U1 + U_];
  }
  __syncthreads();
  if (b == 0) out[0] = -0.25f * (llv[0] + llv[1] + llv[2] + llv[3]);
}

extern "C" void kernel_launch(void* const* d_in, const int* in_sizes, int n_in,
                              void* d_out, int out_size, void* d_ws, size_t ws_size,
                              hipStream_t stream) {
  (void)in_sizes; (void)n_in; (void)out_size; (void)ws_size;
  const float* enc    = (const float*)d_in[0];
  const int*   tokens = (const int*)d_in[1];
  const float* emb    = (const float*)d_in[2];
  const float* W_ih   = (const float*)d_in[3];
  const float* W_hh   = (const float*)d_in[4];
  const float* b_lstm = (const float*)d_in[5];
  const float* W_pred = (const float*)d_in[6];
  const float* b_pred = (const float*)d_in[7];
  const float* W_join = (const float*)d_in[8];
  const float* b_join = (const float*)d_in[9];
  float* out = (float*)d_out;

  char* ws = (char*)d_ws;
  size_t off = 0;
  auto take = [&](size_t bytes) { char* p = ws + off; off = (off + bytes + 255) & ~(size_t)255; return p; };
  unsigned short* xemb    = (unsigned short*)take((size_t)16 * U1 * D_ * 2);   // padded to 16 rows/b
  unsigned short* Wih_b   = (unsigned short*)take((size_t)4 * H_ * D_ * 2);
  unsigned short* Whh_b   = (unsigned short*)take((size_t)4 * H_ * H_ * 2);
  unsigned short* Wpred_b = (unsigned short*)take((size_t)E_ * H_ * 2);
  unsigned short* Wjoin_b = (unsigned short*)take((size_t)V_ * E_ * 2);
  unsigned short* hstate  = (unsigned short*)take((size_t)ROWS_PAD * H_ * 2);  // padded rows
  float* pred_out         = (float*)take((size_t)ROWS * E_ * 4);
  float* blank_lp         = (float*)take((size_t)B_ * T_ * U1 * 4);
  float* label_lp         = (float*)take((size_t)B_ * T_ * U_ * 4);

  auto cvt = [&](const float* s, unsigned short* d, int n) {
    f2bf_kernel<<<(n + 255) / 256, 256, 0, stream>>>(s, d, n);
  };
  cvt(W_ih,   Wih_b,   4 * H_ * D_);
  cvt(W_hh,   Whh_b,   4 * H_ * H_);
  cvt(W_pred, Wpred_b, E_ * H_);
  cvt(W_join, Wjoin_b, V_ * E_);
  embed_kernel<<<(16 * U1 * D_ + 255) / 256, 256, 0, stream>>>(tokens, emb, xemb);
  zerou16_kernel<<<((ROWS_PAD - ROWS) * H_ + 255) / 256, 256, 0, stream>>>(
      hstate + (size_t)ROWS * H_, (ROWS_PAD - ROWS) * H_);
  lstm_kernel<<<1, 256, 0, stream>>>(xemb, Wih_b, Whh_b, b_lstm, hstate);
  pred_kernel<<<26, 256, 0, stream>>>(hstate, Wpred_b, b_pred, pred_out);
  joiner_kernel<<<dim3(7, T_, B_), 256, 0, stream>>>(enc, pred_out, Wjoin_b, b_join, tokens,
                                                     blank_lp, label_lp);
  dp_kernel<<<1, 32, 0, stream>>>(blank_lp, label_lp, out);
}
